// GCN_MIL_MultiClass_10668698763766
// MI455X (gfx1250) — matrix operand order repaired and checked by reading, not compiled, not run on hardware
//
#include <hip/hip_runtime.h>
#include <hip/hip_bf16.h>
#include <math.h>

// ---------------- problem constants (match reference) ----------------
#define BGRAPH 8
#define NPG    4096
#define NNODES (BGRAPH * NPG)        // 32768
#define DEG    8
#define NEDGES (NNODES * DEG)        // 262144
#define IN_C   1024
#define HID    512
#define NCLS   4
#define K1     (NPG / 2)             // 2048
#define K2     (NPG / 4)             // 1024

typedef __attribute__((ext_vector_type(16))) __bf16 v16bf;
typedef __attribute__((ext_vector_type(8)))  float  v8f;

// ---------------- utility ----------------
__global__ void fill_kernel(float* __restrict__ p, float v, int n) {
  int i = blockIdx.x * blockDim.x + threadIdx.x;
  if (i < n) p[i] = v;
}

// f32 -> bf16, 4 elements/thread
__global__ void cvt_bf16_kernel(const float* __restrict__ in,
                                __bf16* __restrict__ out, int n4) {
  int i = blockIdx.x * blockDim.x + threadIdx.x;
  if (i < n4) {
    float4 v = ((const float4*)in)[i];
    __bf16 o[4] = {(__bf16)v.x, (__bf16)v.y, (__bf16)v.z, (__bf16)v.w};
    *(uint2*)(out + (size_t)i * 4) = *(const uint2*)o;
  }
}

// Wt[n][k] = (bf16) W[k][n]   (one-time, tiny)
__global__ void transpose_cvt_kernel(const float* __restrict__ W,
                                     __bf16* __restrict__ Wt, int K, int Nc) {
  int k = blockIdx.x * blockDim.x + threadIdx.x;
  int n = blockIdx.y;
  if (k < K) Wt[(size_t)n * K + k] = (__bf16)W[(size_t)k * Nc + n];
}

// ---------------- bf16 WMMA GEMM: C[M,Nc] = A[M,K] @ Bt[Nc,K]^T ------------
// A, Bt are bf16, K-contiguous.  Tile 128x64x32, 256 threads = 8 waves
// (4 along M x 2 along N); each wave owns a 32x32 sub-tile = 2x2
// v_wmma_f32_16x16x32_bf16 fragments.  Tiles are double-buffered in LDS and
// staged with GLOBAL_LOAD_ASYNC_TO_LDS_B128 (ASYNCcnt) -- the compute stream
// only pays one s_wait_asynccnt 0 before each tile-swap barrier.
#define BM 128
#define BN 64
#define BK 32
#define LDP (BK + 8)   // padded K-stride: 40 bf16 = 80B (16B-aligned, 20 dwords)

__device__ __forceinline__ void async_cp16(const __bf16* lds_dst,
                                           const __bf16* gsrc) {
  unsigned loff = (unsigned)(size_t)lds_dst;   // low 32 bits = LDS address
  asm volatile("global_load_async_to_lds_b128 %0, %1, off"
               :: "v"(loff), "v"(gsrc)
               : "memory");
}
__device__ __forceinline__ void wait_async0() {
  asm volatile("s_wait_asynccnt 0x0" ::: "memory");
}

__global__ __launch_bounds__(256) void gemm_bf16_wmma_kernel(
    const __bf16* __restrict__ A, const __bf16* __restrict__ Bt,
    float* __restrict__ C, int M, int K, int Nc) {
  __shared__ __bf16 As[2][BM][LDP];   // [row][k]
  __shared__ __bf16 Bs[2][BN][LDP];   // [col][k]  (B pre-transposed in global)

  const int tid  = threadIdx.x;
  const int lane = tid & 31;
  const int wave = tid >> 5;       // 0..7
  const int wm   = wave & 3;       // 4 waves along M
  const int wn   = wave >> 2;      // 2 waves along N
  const int bm   = blockIdx.x * BM;
  const int bn   = blockIdx.y * BN;
  const int half = lane >> 4;      // K-half selector per ISA layout
  const int l16  = lane & 15;

  // staging units: one 16B (8 x bf16) chunk per unit
  const int ar0 = tid >> 2;        // A rows: unit u -> row u>>2, quad u&3
  const int aq0 = tid & 3;
  const int br  = tid >> 2;        // B: 64 cols x 4 quads = 256 units
  const int bq  = tid & 3;

  v8f acc[2][2] = {};

  auto stage = [&](int buf, int kt) {
    // A tile: 128 rows x 4 quads = 512 units, 2 per thread
    async_cp16(&As[buf][ar0][aq0 * 8],
               A + (size_t)(bm + ar0) * K + kt + aq0 * 8);
    async_cp16(&As[buf][ar0 + 64][aq0 * 8],
               A + (size_t)(bm + ar0 + 64) * K + kt + aq0 * 8);
    // B tile: 64 cols x 4 quads = 256 units, 1 per thread
    async_cp16(&Bs[buf][br][bq * 8],
               Bt + (size_t)(bn + br) * K + kt + bq * 8);
  };

  stage(0, 0);
  wait_async0();
  __syncthreads();
  int buf = 0;

  for (int kt = 0; kt < K; kt += BK) {
    if (kt + BK < K) stage(buf ^ 1, kt + BK);  // async fill of idle buffer

    // A fragments: lane 0-15 -> K 0..7 / 16..23 ; lane 16-31 -> K 8..15 / 24..31
    v16bf afrag[2];
#pragma unroll
    for (int mi = 0; mi < 2; ++mi) {
      int row = wm * 32 + mi * 16 + l16;
#pragma unroll
      for (int j = 0; j < 8; ++j) {
        afrag[mi][j]     = As[buf][row][half * 8 + j];
        afrag[mi][j + 8] = As[buf][row][16 + half * 8 + j];
      }
    }
    // B fragments: lane 0-15 -> K 0..7 / 16..23 ; lane 16-31 -> K 8..15 / 24..31
    v16bf bfrag[2];
#pragma unroll
    for (int ni = 0; ni < 2; ++ni) {
      int col = wn * 32 + ni * 16 + l16;
#pragma unroll
      for (int j = 0; j < 16; ++j)
        bfrag[ni][j] = Bs[buf][col][j < 8 ? half * 8 + j : 16 + half * 8 + (j - 8)];
    }

#pragma unroll
    for (int mi = 0; mi < 2; ++mi)
#pragma unroll
      for (int ni = 0; ni < 2; ++ni)
        acc[mi][ni] = __builtin_amdgcn_wmma_f32_16x16x32_bf16(
            false, afrag[mi], false, bfrag[ni], (short)0, acc[mi][ni],
            false, false);

    wait_async0();                 // my async copies have landed in LDS
    __syncthreads();               // everyone's copies + compute done
    buf ^= 1;
  }

  // store: C/D layout, VGPR r -> M = r (lanes 0-15) / 8+r (lanes 16-31), N = lane%16
#pragma unroll
  for (int mi = 0; mi < 2; ++mi)
#pragma unroll
    for (int ni = 0; ni < 2; ++ni) {
#pragma unroll
      for (int r = 0; r < 8; ++r) {
        int row = bm + wm * 32 + mi * 16 + half * 8 + r;
        int col = bn + wn * 32 + ni * 16 + l16;
        C[(size_t)row * Nc + col] = acc[mi][ni][r];
      }
    }
}

// ---------------- GCN aggregation ----------------
__global__ void edge_degree_kernel(const int* __restrict__ dst,
                                   const float* __restrict__ emask,
                                   float* __restrict__ deg, int E) {
  int e = blockIdx.x * blockDim.x + threadIdx.x;
  if (e < E) {
    float m = emask[e];
    if (m != 0.f) atomicAdd(&deg[dst[e]], m);
  }
}

__global__ void dinv_kernel(const float* __restrict__ deg,
                            const float* __restrict__ nmask,
                            float* __restrict__ dinv, int Nn) {
  int n = blockIdx.x * blockDim.x + threadIdx.x;
  if (n < Nn) {
    float d = deg[n] + nmask[n];
    dinv[n] = (d > 0.f) ? rsqrtf(d) : 0.f;
  }
}

// one block (128 thr, float4 lanes) per edge; skip masked edges early
__global__ __launch_bounds__(128) void aggregate_kernel(
    const float* __restrict__ h, const int* __restrict__ src,
    const int* __restrict__ dst, const float* __restrict__ emask,
    const float* __restrict__ dinv, float* __restrict__ agg) {
  int e = blockIdx.x;
  float em = emask[e];
  if (em <= 0.f) return;
  int s = src[e], d = dst[e];
  float coef = dinv[s] * dinv[d] * em;
  if (coef == 0.f) return;
  const float4* hs = (const float4*)(h + (size_t)s * HID);
  float* ad = agg + (size_t)d * HID;
  int c = threadIdx.x;                 // 128 threads x float4 = 512 channels
  float4 v = hs[c];
  atomicAdd(&ad[c * 4 + 0], v.x * coef);
  atomicAdd(&ad[c * 4 + 1], v.y * coef);
  atomicAdd(&ad[c * 4 + 2], v.z * coef);
  atomicAdd(&ad[c * 4 + 3], v.w * coef);
}

// out = lrelu(agg + dinv^2*nmask*h + b)  -> H
__global__ __launch_bounds__(128) void finish_gcn_kernel(
    const float* __restrict__ agg, const float* __restrict__ g,
    const float* __restrict__ dinv, const float* __restrict__ nmask,
    const float* __restrict__ bias, float* __restrict__ H) {
  int n = blockIdx.x;
  float self = dinv[n] * dinv[n] * nmask[n];
  int c = threadIdx.x;                 // 128 threads x float4
  const float4* ap = (const float4*)(agg + (size_t)n * HID);
  const float4* gp = (const float4*)(g + (size_t)n * HID);
  const float4* bp = (const float4*)bias;
  float4 a = ap[c], hv = gp[c], b4 = bp[c];
  float4 o;
  o.x = a.x + self * hv.x + b4.x;
  o.y = a.y + self * hv.y + b4.y;
  o.z = a.z + self * hv.z + b4.z;
  o.w = a.w + self * hv.w + b4.w;
  o.x = o.x > 0.f ? o.x : 0.01f * o.x;
  o.y = o.y > 0.f ? o.y : 0.01f * o.y;
  o.z = o.z > 0.f ? o.z : 0.01f * o.z;
  o.w = o.w > 0.f ? o.w : 0.01f * o.w;
  ((float4*)(H + (size_t)n * HID))[c] = o;
}

// ---------------- SAG pooling ----------------
// per-node dots: t[n] = h[n].wr ; score[n] = h[n].wo + br  (edge sum added later)
__global__ __launch_bounds__(128) void score_init_kernel(
    const float* __restrict__ h, const float* __restrict__ wr,
    const float* __restrict__ wo, const float* __restrict__ br,
    float* __restrict__ tvec, float* __restrict__ score) {
  __shared__ float sr[128], so[128];
  int n = blockIdx.x, tid = threadIdx.x;
  const float* hp = h + (size_t)n * HID;
  float tr = 0.f, to = 0.f;
  for (int c = tid; c < HID; c += 128) {
    float v = hp[c];
    tr += v * wr[c];
    to += v * wo[c];
  }
  sr[tid] = tr; so[tid] = to;
  __syncthreads();
  for (int s2 = 64; s2 > 0; s2 >>= 1) {
    if (tid < s2) { sr[tid] += sr[tid + s2]; so[tid] += so[tid + s2]; }
    __syncthreads();
  }
  if (tid == 0) { tvec[n] = sr[0]; score[n] = so[0] + br[0]; }
}

__global__ void score_edge_kernel(const float* __restrict__ tvec,
                                  const int* __restrict__ src,
                                  const int* __restrict__ dst,
                                  const float* __restrict__ emask,
                                  float* __restrict__ score, int E) {
  int e = blockIdx.x * blockDim.x + threadIdx.x;
  if (e < E) {
    float m = emask[e];
    if (m > 0.f) atomicAdd(&score[dst[e]], tvec[src[e]] * m);
  }
}

// rank-by-counting top-k (stable argsort(-score) semantics); grid (B, NPG/256)
__global__ __launch_bounds__(256) void topk_kernel(
    const float* __restrict__ score, const float* __restrict__ nmask,
    float* __restrict__ keep, int k) {
  __shared__ float s[NPG];
  int g = blockIdx.x;
  int base = g * NPG;
  for (int i = threadIdx.x; i < NPG; i += 256) {
    float m = nmask[base + i];
    s[i] = (m > 0.f) ? score[base + i] : -1e30f;
  }
  __syncthreads();
  int li = blockIdx.y * 256 + threadIdx.x;
  float my = s[li];
  int rank = 0;
  for (int j = 0; j < NPG; ++j) {
    float sj = s[j];
    rank += (sj > my) || (sj == my && j < li);
  }
  keep[base + li] = (rank < k) ? nmask[base + li] : 0.f;
}

// h *= tanh(score)*keep ; nmask = keep ; also emit bf16 copy for next GEMM
__global__ __launch_bounds__(128) void pool_apply_kernel(
    float* __restrict__ h, __bf16* __restrict__ hb,
    const float* __restrict__ score, const float* __restrict__ keep,
    float* __restrict__ nmask) {
  int n = blockIdx.x;
  float kp = keep[n];
  float sc = tanhf(score[n]) * kp;
  float4* hp = (float4*)(h + (size_t)n * HID);
  int c = threadIdx.x;
  float4 v = hp[c];
  v.x *= sc; v.y *= sc; v.z *= sc; v.w *= sc;
  hp[c] = v;
  __bf16 o[4] = {(__bf16)v.x, (__bf16)v.y, (__bf16)v.z, (__bf16)v.w};
  ((uint2*)(hb + (size_t)n * HID))[c] = *(const uint2*)o;
  if (threadIdx.x == 0) nmask[n] = kp;
}

__global__ void edge_update_kernel(float* __restrict__ emask,
                                   const float* __restrict__ keep,
                                   const int* __restrict__ src,
                                   const int* __restrict__ dst, int E) {
  int e = blockIdx.x * blockDim.x + threadIdx.x;
  if (e < E) emask[e] *= keep[src[e]] * keep[dst[e]];
}

// ---------------- readout + MLP ----------------
__global__ void mask_sum_kernel(const float* __restrict__ nmask,
                                float* __restrict__ msum, int Nn) {
  int n = blockIdx.x * blockDim.x + threadIdx.x;
  if (n < Nn) atomicAdd(&msum[n / NPG], nmask[n]);
}

// grid (B, HID/256): gcat[b] = [mean | max]
__global__ __launch_bounds__(256) void readout_kernel(
    const float* __restrict__ h, const float* __restrict__ nmask,
    const float* __restrict__ msum, float* __restrict__ gcat) {
  int b = blockIdx.x;
  int c = blockIdx.y * 256 + threadIdx.x;
  const float* hb = h + (size_t)b * NPG * HID;
  const float* mb = nmask + b * NPG;
  float sum = 0.f, mx = -INFINITY;
  for (int n = 0; n < NPG; ++n) {
    float m = mb[n];
    float v = hb[(size_t)n * HID + c];
    sum += v * m;
    if (m > 0.f) mx = fmaxf(mx, v);
  }
  gcat[b * (2 * HID) + c] = sum / msum[b];
  gcat[b * (2 * HID) + HID + c] = mx;
}

__global__ __launch_bounds__(512) void fc1_kernel(
    const float* __restrict__ gcat, const float* __restrict__ w,
    const float* __restrict__ bias, float* __restrict__ out) {
  int b = blockIdx.x, j = threadIdx.x;
  float acc = bias[j];
  const float* g = gcat + b * (2 * HID);
  for (int k = 0; k < 2 * HID; ++k) acc += g[k] * w[k * HID + j];
  out[b * HID + j] = acc > 0.f ? acc : 0.01f * acc;
}

__global__ void fc2_kernel(const float* __restrict__ in,
                           const float* __restrict__ w,
                           const float* __restrict__ bias,
                           float* __restrict__ out) {
  int t = threadIdx.x;
  if (t >= BGRAPH * NCLS) return;
  int b = t >> 2, c = t & 3;
  float acc = bias[c];
  for (int j = 0; j < HID; ++j) acc += in[b * HID + j] * w[j * NCLS + c];
  out[b * NCLS + c] = acc;
}

// ---------------- host orchestration ----------------
static void run_gcn_layer(const __bf16* Ab, int Kdim, const __bf16* Wt,
                          const float* bias, const int* src, const int* dst,
                          const float* emask, const float* nmask, float* G,
                          float* AGG, float* deg, float* dinv, float* H,
                          hipStream_t stream) {
  dim3 gg(NNODES / BM, HID / BN);
  gemm_bf16_wmma_kernel<<<gg, 256, 0, stream>>>(Ab, Wt, G, NNODES, Kdim, HID);
  fill_kernel<<<(NNODES + 255) / 256, 256, 0, stream>>>(deg, 0.f, NNODES);
  edge_degree_kernel<<<(NEDGES + 255) / 256, 256, 0, stream>>>(dst, emask, deg, NEDGES);
  dinv_kernel<<<(NNODES + 255) / 256, 256, 0, stream>>>(deg, nmask, dinv, NNODES);
  int agg_elems = NNODES * HID;
  fill_kernel<<<(agg_elems + 255) / 256, 256, 0, stream>>>(AGG, 0.f, agg_elems);
  aggregate_kernel<<<NEDGES, 128, 0, stream>>>(G, src, dst, emask, dinv, AGG);
  finish_gcn_kernel<<<NNODES, 128, 0, stream>>>(AGG, G, dinv, nmask, bias, H);
}

static void run_sag_pool(float* H, __bf16* Hb, const float* wr, const float* br,
                         const float* wo, int k, const int* src, const int* dst,
                         float* emask, float* nmask, float* tvec, float* score,
                         float* keep, hipStream_t stream) {
  score_init_kernel<<<NNODES, 128, 0, stream>>>(H, wr, wo, br, tvec, score);
  score_edge_kernel<<<(NEDGES + 255) / 256, 256, 0, stream>>>(tvec, src, dst, emask, score, NEDGES);
  topk_kernel<<<dim3(BGRAPH, NPG / 256), 256, 0, stream>>>(score, nmask, keep, k);
  pool_apply_kernel<<<NNODES, 128, 0, stream>>>(H, Hb, score, keep, nmask);
  edge_update_kernel<<<(NEDGES + 255) / 256, 256, 0, stream>>>(emask, keep, src, dst, NEDGES);
}

extern "C" void kernel_launch(void* const* d_in, const int* in_sizes, int n_in,
                              void* d_out, int out_size, void* d_ws, size_t ws_size,
                              hipStream_t stream) {
  // inputs per setup_inputs() order
  const float* x     = (const float*)d_in[0];
  const int*   src   = (const int*)d_in[1];
  const int*   dst   = (const int*)d_in[2];
  const float* W1    = (const float*)d_in[3];
  const float* b1    = (const float*)d_in[4];
  const float* p1_wr = (const float*)d_in[5];
  const float* p1_br = (const float*)d_in[6];
  const float* p1_wo = (const float*)d_in[7];
  const float* W2    = (const float*)d_in[8];
  const float* b2    = (const float*)d_in[9];
  const float* p2_wr = (const float*)d_in[10];
  const float* p2_br = (const float*)d_in[11];
  const float* p2_wo = (const float*)d_in[12];
  const float* W3    = (const float*)d_in[13];
  const float* b3    = (const float*)d_in[14];
  const float* fc1_w = (const float*)d_in[15];
  const float* fc1_b = (const float*)d_in[16];
  const float* fc2_w = (const float*)d_in[17];
  const float* fc2_b = (const float*)d_in[18];
  float* out = (float*)d_out;

  // workspace carving
  size_t off = 0;
  auto carve = [&](size_t bytes) -> void* {
    void* p = (char*)d_ws + off;
    off = (off + bytes + 255) & ~(size_t)255;
    return p;
  };
  float*  G     = (float*)carve((size_t)NNODES * HID * 4);    // GEMM output
  float*  AGG   = (float*)carve((size_t)NNODES * IN_C * 2);   // agg accum; aliased with xb
  __bf16* xb    = (__bf16*)AGG;                               // xb dead before AGG is used
  float*  H     = (float*)carve((size_t)NNODES * HID * 4);    // node features fp32
  __bf16* Hb    = (__bf16*)carve((size_t)NNODES * HID * 2);   // bf16 copy for next GEMM
  __bf16* W1t   = (__bf16*)carve((size_t)HID * IN_C * 2);     // transposed bf16 weights
  __bf16* W2t   = (__bf16*)carve((size_t)HID * HID * 2);
  __bf16* W3t   = (__bf16*)carve((size_t)HID * HID * 2);
  float*  deg   = (float*)carve((size_t)NNODES * 4);
  float*  dinv  = (float*)carve((size_t)NNODES * 4);
  float*  tvec  = (float*)carve((size_t)NNODES * 4);
  float*  score = (float*)carve((size_t)NNODES * 4);
  float*  keep  = (float*)carve((size_t)NNODES * 4);
  float*  nmask = (float*)carve((size_t)NNODES * 4);
  float*  emask = (float*)carve((size_t)NEDGES * 4);
  float*  msum  = (float*)carve((size_t)BGRAPH * 4);
  float*  gcat  = (float*)carve((size_t)BGRAPH * 2 * HID * 4);
  float*  fc1o  = (float*)carve((size_t)BGRAPH * HID * 4);
  (void)ws_size; (void)n_in; (void)in_sizes; (void)out_size;

  // one-time conversions (run every launch; deterministic)
  cvt_bf16_kernel<<<(NNODES * IN_C / 4 + 255) / 256, 256, 0, stream>>>(x, xb, NNODES * IN_C / 4);
  transpose_cvt_kernel<<<dim3(IN_C / 256, HID), 256, 0, stream>>>(W1, W1t, IN_C, HID);
  transpose_cvt_kernel<<<dim3(HID / 256, HID), 256, 0, stream>>>(W2, W2t, HID, HID);
  transpose_cvt_kernel<<<dim3(HID / 256, HID), 256, 0, stream>>>(W3, W3t, HID, HID);

  // init masks
  fill_kernel<<<(NEDGES + 255) / 256, 256, 0, stream>>>(emask, 1.f, NEDGES);
  fill_kernel<<<(NNODES + 255) / 256, 256, 0, stream>>>(nmask, 1.f, NNODES);

  // layer 1 + pool 1   (AGG is zero-filled only after GEMM-1 consumed xb)
  run_gcn_layer(xb, IN_C, W1t, b1, src, dst, emask, nmask, G, AGG, deg, dinv, H, stream);
  run_sag_pool(H, Hb, p1_wr, p1_br, p1_wo, K1, src, dst, emask, nmask, tvec, score, keep, stream);

  // layer 2 + pool 2
  run_gcn_layer(Hb, HID, W2t, b2, src, dst, emask, nmask, G, AGG, deg, dinv, H, stream);
  run_sag_pool(H, Hb, p2_wr, p2_br, p2_wo, K2, src, dst, emask, nmask, tvec, score, keep, stream);

  // layer 3
  run_gcn_layer(Hb, HID, W3t, b3, src, dst, emask, nmask, G, AGG, deg, dinv, H, stream);

  // readout + MLP
  fill_kernel<<<1, 256, 0, stream>>>(msum, 0.f, BGRAPH);
  mask_sum_kernel<<<(NNODES + 255) / 256, 256, 0, stream>>>(nmask, msum, NNODES);
  readout_kernel<<<dim3(BGRAPH, HID / 256), 256, 0, stream>>>(H, nmask, msum, gcat);
  fc1_kernel<<<BGRAPH, HID, 0, stream>>>(gcat, fc1_w, fc1_b, fc1o);
  fc2_kernel<<<1, 32, 0, stream>>>(fc1o, fc2_w, fc2_b, out);
}
